// StackLSTM_67654324847049
// MI455X (gfx1250) — matrix operand order repaired and checked by reading, not compiled, not run on hardware
//
#include <hip/hip_runtime.h>
#include <hip/hip_bf16.h>

// ---------------------------------------------------------------------------
// Stack-LSTM (RNNG) for MI455X / gfx1250, wave32 + WMMA f16.
//
//  * grid = 4 blocks x 512 threads (16 waves); each block owns 16 batch items
//    for the whole T=256 step recurrence (items are independent).
//  * every matmul is a 16xN GEMM mapped to v_wmma_f32_16x16x32_f16:
//      gates(16x1024) = X(16x256) @ W_ih^T + H(16x256) @ W_hh^T
//    wave w owns the 4 gate tiles {i,f,g,o} of hidden columns [16w,16w+16),
//    so cell state c lives in v8f accumulator registers across the recurrence.
//  * weights are repacked once per call into B-fragment order (f16) so each
//    B fragment is one coalesced 32-byte load per lane.
//  * child stack S (64 items x 64 slots x 256) lives in d_ws as f16 (~2MB);
//    total d_ws use ~5.3MB (packed weights 3.25MB + biases + S).
// ---------------------------------------------------------------------------

typedef _Float16 half_t;
typedef __attribute__((ext_vector_type(16))) _Float16 v16h;
typedef __attribute__((ext_vector_type(8)))  _Float16 v8h;
typedef __attribute__((ext_vector_type(8)))  float    v8f;

#define T_STEPS 256
#define BATCH   64
#define HS      256          // SIZE
#define MAXK    16
#define MAXN    64
#define MAXF    16
#define GROUP   16           // items per block
#define NBLK    (BATCH / GROUP)
#define BT      512          // threads per block = 16 waves
#define LSTRIDE 264          // padded halves per LDS row (528B, 16B aligned)

// ws layout ---------------------------------------------------------------
#define H_BIG       262144u          // halves per 1024x256 matrix
#define OFF_WC_H    (6u * H_BIG)     // Wc packed starts here (halves)
#define WC_HALVES   131072u
#define OFF_BIAS_B  3407872u         // bytes: fused biases (3*1024 + 256 f32)
#define OFF_S_B     3421184u         // bytes: stack S, 64*64*256 f16 = 2MB
// total ws required: 5,518,336 bytes (~5.3MB)

// ---------------------------------------------------------------------------
__device__ __forceinline__ v8f zero_v8f() {
  v8f z;
#pragma unroll
  for (int i = 0; i < 8; ++i) z[i] = 0.0f;
  return z;
}

__device__ __forceinline__ float sigf(float x) { return 1.0f / (1.0f + __expf(-x)); }

// A fragment (16x32 f16, M x K) from an LDS tile with row stride LSTRIDE.
// Layout: lanes 0-15 -> M=lane, K=0..15 contiguous;
//         lanes 16-31 -> K=16..31 (matches packed B order).
__device__ __forceinline__ v16h load_a_frag(const half_t* base, int lane) {
  const int m  = lane & 15;
  const int kb = (lane & 16) ? 16 : 0;
  const half_t* r = base + m * LSTRIDE + kb;
  v8h lo = *reinterpret_cast<const v8h*>(r);        // K = kb .. kb+7
  v8h hi = *reinterpret_cast<const v8h*>(r + 8);    // K = kb+8 .. kb+15
  v16h a;
#pragma unroll
  for (int i = 0; i < 8; ++i) { a[i] = lo[i]; a[i + 8] = hi[i]; }
  return a;
}

// B fragment: pre-packed lane-major, one contiguous 32B load per lane.
__device__ __forceinline__ v16h load_b_frag(const half_t* packed, int tile, int lane) {
  return *reinterpret_cast<const v16h*>(packed + ((size_t)tile * 32 + lane) * 16);
}

__device__ __forceinline__ v8f wmma_f16(v16h a, v16h b, v8f c) {
  return __builtin_amdgcn_wmma_f32_16x16x32_f16(false, a, false, b, (short)0, c,
                                                false, false);
}

// gates = X @ Wx^T + H @ Wh^T  (wave w -> gate tiles g*16+w, g=0..3; K=256)
__device__ __forceinline__ void gemm_gates(v8f acc[4], const half_t* sXb,
                                           const half_t* sHb, const half_t* Wx,
                                           const half_t* Wh, int w, int lane) {
#pragma unroll
  for (int g = 0; g < 4; ++g) {
    v8f a = acc[g];
    const int ntile = (g << 4) + w;
#pragma unroll
    for (int kk = 0; kk < 8; ++kk)
      a = wmma_f16(load_a_frag(sXb + kk * 32, lane),
                   load_b_frag(Wx, ntile * 8 + kk, lane), a);
#pragma unroll
    for (int kk = 0; kk < 8; ++kk)
      a = wmma_f16(load_a_frag(sHb + kk * 32, lane),
                   load_b_frag(Wh, ntile * 8 + kk, lane), a);
    acc[g] = a;
  }
}

// one chained stack-LSTM cell step for the whole 16-item group
__device__ __forceinline__ void main_cell_step(
    const half_t* sX, half_t* sHm, const half_t* WIHp, const half_t* WHHp,
    const float bM[4], v8f& cmain, float hmain[8], const int* validArr,
    int w, int lane, float* outRow) {
  v8f acc[4];
#pragma unroll
  for (int g = 0; g < 4; ++g) acc[g] = zero_v8f();
  gemm_gates(acc, sX, sHm, WIHp, WHHp, w, lane);
  __syncthreads();                         // all reads of sHm done before writes
  const int colw = (w << 4) + (lane & 15);
  const int mhi  = (lane & 16) ? 8 : 0;
#pragma unroll
  for (int v = 0; v < 8; ++v) {
    const int m = v + mhi;
    if (validArr[m]) {
      float ig = sigf(acc[0][v] + bM[0]);
      float fg = sigf(acc[1][v] + bM[1]);
      float gg = tanhf(acc[2][v] + bM[2]);
      float og = sigf(acc[3][v] + bM[3]);
      float c2 = fg * cmain[v] + ig * gg;
      cmain[v] = c2;
      hmain[v] = og * tanhf(c2);
    }
    sHm[m * LSTRIDE + colw]     = (half_t)hmain[v];
    outRow[(size_t)m * HS + colw] = hmain[v];
  }
  __syncthreads();
}

// ---------------------------------------------------------------------------
// weight repack: f32 row-major (N=1024|256, K=256|512) -> f16 B-fragment order
__global__ void pack_weights(const float* __restrict__ Wih, const float* __restrict__ Whh,
                             const float* __restrict__ Wfih, const float* __restrict__ Wfhh,
                             const float* __restrict__ Wbih, const float* __restrict__ Wbhh,
                             const float* __restrict__ Wc, half_t* __restrict__ dst) {
  const unsigned idx = blockIdx.x * 256u + threadIdx.x;
  if (idx < 6u * H_BIG) {
    const unsigned mi = idx / H_BIG, rem = idx % H_BIG;
    const float* src = (mi == 0) ? Wih : (mi == 1) ? Whh : (mi == 2) ? Wfih
                     : (mi == 3) ? Wfhh : (mi == 4) ? Wbih : Wbhh;
    const int ntile = rem >> 12;             // / (8*32*16)
    const int kk    = (rem >> 9) & 7;
    const int lane  = (rem >> 4) & 31;
    const int i     = rem & 15;
    const int n = (ntile << 4) + (lane & 15);
    const int k = (kk << 5) + ((lane & 16) ? 16 : 0) + i;
    dst[idx] = (half_t)src[n * 256 + k];
  } else if (idx < 6u * H_BIG + WC_HALVES) {
    const unsigned j = idx - 6u * H_BIG;
    const int ntile = j >> 13;               // / (16*32*16)
    const int kk    = (j >> 9) & 15;
    const int lane  = (j >> 4) & 31;
    const int i     = j & 15;
    const int n = (ntile << 4) + (lane & 15);
    const int k = (kk << 5) + ((lane & 16) ? 16 : 0) + i;
    dst[idx] = (half_t)Wc[n * 512 + k];
  }
}

__global__ void prep_bias(const float* __restrict__ bih, const float* __restrict__ bhh,
                          const float* __restrict__ bfih, const float* __restrict__ bfhh,
                          const float* __restrict__ bbih, const float* __restrict__ bbhh,
                          const float* __restrict__ bc, float* __restrict__ dst) {
  const int i = blockIdx.x * 256 + threadIdx.x;
  if (i < 1024)       dst[i] = bih[i] + bhh[i];
  else if (i < 2048)  dst[i] = bfih[i - 1024] + bfhh[i - 1024];
  else if (i < 3072)  dst[i] = bbih[i - 2048] + bbhh[i - 2048];
  else if (i < 3328)  dst[i] = bc[i - 3072];
}

// ---------------------------------------------------------------------------
__global__ __launch_bounds__(BT)
void stack_lstm_kernel(const float* __restrict__ xs, const int* __restrict__ actions,
                       const float* __restrict__ open_wv, char* __restrict__ wsb,
                       float* __restrict__ out) {
  __shared__ alignas(16) half_t sX  [GROUP * LSTRIDE];  // cell input (A operand)
  __shared__ alignas(16) half_t sH  [GROUP * LSTRIDE];  // inner-LSTM h
  __shared__ alignas(16) half_t sHf [GROUP * LSTRIDE];  // fwd result
  __shared__ alignas(16) half_t sHbw[GROUP * LSTRIDE];  // bwd result
  __shared__ alignas(16) half_t sHm [GROUP * LSTRIDE];  // main-cell h (f16 copy)
  __shared__ alignas(16) half_t sXc [GROUP * LSTRIDE];  // composed x
  __shared__ int sp[GROUP], sd[GROUP], sfT[GROUP], sfs[GROUP], sL[GROUP];
  __shared__ int sIsSh[GROUP], sIsRd[GROUP], sIsOp[GROUP], sValid[GROUP], sWidx[GROUP];
  __shared__ int sF[GROUP][MAXF];

  const int tid   = threadIdx.x;
  const int lane  = tid & 31;
  const int w     = tid >> 5;                 // wave id 0..15
  const int item0 = blockIdx.x * GROUP;

  const half_t* Wp    = (const half_t*)wsb;
  const half_t* WIHp  = Wp;
  const half_t* WHHp  = Wp + 1 * H_BIG;
  const half_t* WFIHp = Wp + 2 * H_BIG;
  const half_t* WFHHp = Wp + 3 * H_BIG;
  const half_t* WBIHp = Wp + 4 * H_BIG;
  const half_t* WBHHp = Wp + 5 * H_BIG;
  const half_t* WCp   = Wp + OFF_WC_H;
  const float*  biases = (const float*)(wsb + OFF_BIAS_B);
  half_t* Sg   = (half_t*)(wsb + OFF_S_B);
  half_t* Sblk = Sg + (size_t)item0 * MAXN * HS;

  const int nb = (w << 4) + (lane & 15);      // this lane's hidden column
  float bMr[4], bFr[4], bBr[4];
#pragma unroll
  for (int g = 0; g < 4; ++g) {
    bMr[g] = biases[g * 256 + nb];
    bFr[g] = biases[1024 + g * 256 + nb];
    bBr[g] = biases[2048 + g * 256 + nb];
  }
  const float bCr = biases[3072 + nb];

  // init: zero stack region, h state; stage x = open_wv; valid = all
  for (int e = tid; e < GROUP * MAXN * HS; e += BT) Sblk[e] = (half_t)0.0f;
  for (int e = tid; e < GROUP * LSTRIDE; e += BT) sHm[e] = (half_t)0.0f;
  for (int e = tid; e < GROUP * HS; e += BT) {
    const int m = e >> 8, col = e & 255;
    sX[m * LSTRIDE + col] = (half_t)open_wv[col];
  }
  if (tid < GROUP) {
    sp[tid] = 0; sd[tid] = 1; sValid[tid] = 1;
    for (int i = 0; i < MAXF; ++i) sF[tid][i] = 0;
  }
  v8f cmain = zero_v8f();
  float hmain[8];
#pragma unroll
  for (int v = 0; v < 8; ++v) hmain[v] = 0.0f;
  __syncthreads();

  // h0 = cell(open_wv, 0, 0) -> out[0]
  main_cell_step(sX, sHm, WIHp, WHHp, bMr, cmain, hmain, sValid, w, lane,
                 out + (size_t)item0 * HS);

  for (int t = 0; t < T_STEPS; ++t) {
    // -- phase 1: per-item bookkeeping ------------------------------------
    if (tid < GROUP) {
      const int it = tid;
      const int p = sp[it], d = sd[it];
      int dm1 = d - 1; if (dm1 < 0) dm1 = 0;
      const int f = (d > 0) ? sF[it][dm1] : 0;
      int L = p - f; if (L < 0) L = 0; if (L > MAXK) L = MAXK;
      const int a = actions[(size_t)t * BATCH + item0 + it];
      const int is_sh = (a == 1);
      const int is_rd = (a == 2) && (d > 0) && (p > f);
      const int is_op = (a == 0);
      int fs = f; if (fs < 0) fs = 0; if (fs > MAXN - MAXK) fs = MAXN - MAXK;
      sfT[it] = f; sfs[it] = fs; sL[it] = L;
      sIsSh[it] = is_sh; sIsRd[it] = is_rd; sIsOp[it] = is_op;
      sValid[it] = is_sh | is_op | is_rd;
      int widx = is_rd ? f : ((p < MAXN - 1) ? p : MAXN - 1);
      if (widx < 0) widx = 0; if (widx > MAXN - 1) widx = MAXN - 1;
      sWidx[it] = widx;
    }
    __syncthreads();

    // -- phases 2/3: bidirectional child LSTM over stacked children -------
    for (int dir = 0; dir < 2; ++dir) {
      const half_t* Wx = dir ? WBIHp : WFIHp;
      const half_t* Wh = dir ? WBHHp : WFHHp;
      const float*  bD = dir ? bBr : bFr;
      half_t* sHout = dir ? sHbw : sHf;
      for (int e = tid; e < GROUP * LSTRIDE; e += BT) sH[e] = (half_t)0.0f;
      v8f ci = zero_v8f();
      __syncthreads();
      for (int k = 0; k < MAXK; ++k) {
        // stage child x tile (per-item gather from stack, reversed for bwd)
        for (int e = tid; e < GROUP * HS; e += BT) {
          const int m = e >> 8, col = e & 255;
          int slot;
          if (dir) { slot = sL[m] - 1 - k;
                     if (slot < 0) slot = 0; if (slot > MAXK - 1) slot = MAXK - 1; }
          else     { slot = k; }
          sX[m * LSTRIDE + col] =
              Sblk[((size_t)m * MAXN + sfs[m] + slot) * HS + col];
        }
        __syncthreads();
        v8f acc[4];
#pragma unroll
        for (int g = 0; g < 4; ++g) acc[g] = zero_v8f();
        gemm_gates(acc, sX, sH, Wx, Wh, w, lane);
        __syncthreads();                       // reads of sH done before writes
        const int colw = (w << 4) + (lane & 15);
        const int mhi  = (lane & 16) ? 8 : 0;
#pragma unroll
        for (int v = 0; v < 8; ++v) {
          const int m = v + mhi;
          if (k < sL[m]) {                     // masked recurrence (k < L)
            float ig = sigf(acc[0][v] + bD[0]);
            float fg = sigf(acc[1][v] + bD[1]);
            float gg = tanhf(acc[2][v] + bD[2]);
            float og = sigf(acc[3][v] + bD[3]);
            float c2 = fg * ci[v] + ig * gg;
            ci[v] = c2;
            sH[m * LSTRIDE + colw] = (half_t)(og * tanhf(c2));
          }
        }
        __syncthreads();
      }
      for (int e = tid; e < GROUP * HS; e += BT) {
        const int m = e >> 8, col = e & 255;
        sHout[m * LSTRIDE + col] = sH[m * LSTRIDE + col];
      }
      __syncthreads();
    }

    // -- phase 4: x_comp = tanh([hf, hb] @ Wc^T + bc) ---------------------
    {
      v8f xacc = zero_v8f();
#pragma unroll
      for (int kc = 0; kc < 16; ++kc) {
        const half_t* abase = (kc < 8) ? (sHf + kc * 32) : (sHbw + (kc - 8) * 32);
        xacc = wmma_f16(load_a_frag(abase, lane),
                        load_b_frag(WCp, (w << 4) + kc, lane), xacc);
      }
      const int colw = (w << 4) + (lane & 15);
      const int mhi  = (lane & 16) ? 8 : 0;
#pragma unroll
      for (int v = 0; v < 8; ++v) {
        const int m = v + mhi;
        sXc[m * LSTRIDE + colw] = (half_t)tanhf(xacc[v] + bCr);
      }
      __syncthreads();
    }

    // -- phase 5: select x per action; push to stack on shift/reduce ------
    for (int e = tid; e < GROUP * HS; e += BT) {
      const int m = e >> 8, col = e & 255;
      half_t xv;
      if (sIsSh[m])      xv = (half_t)xs[((size_t)t * BATCH + item0 + m) * HS + col];
      else if (sIsRd[m]) xv = sXc[m * LSTRIDE + col];
      else               xv = (half_t)open_wv[col];
      sX[m * LSTRIDE + col] = xv;
      if (sIsSh[m] || sIsRd[m])
        Sblk[((size_t)m * MAXN + sWidx[m]) * HS + col] = xv;
    }
    __syncthreads();

    // -- phase 6: chained stack-LSTM cell -> out[t+1] ---------------------
    main_cell_step(sX, sHm, WIHp, WHHp, bMr, cmain, hmain, sValid, w, lane,
                   out + ((size_t)(t + 1) * BATCH + item0) * HS);

    // -- phase 7: pointer / depth updates ---------------------------------
    if (tid < GROUP) {
      const int it = tid;
      const int p = sp[it], d = sd[it];
      if (sIsSh[it])      sp[it] = p + 1;
      else if (sIsRd[it]) sp[it] = sfT[it] + 1;
      if (sIsOp[it] && d < MAXF) { sF[it][d] = p; sd[it] = d + 1; }
      else if (sIsRd[it])        { sd[it] = d - 1; }
    }
    __syncthreads();
  }
}

// ---------------------------------------------------------------------------
extern "C" void kernel_launch(void* const* d_in, const int* in_sizes, int n_in,
                              void* d_out, int out_size, void* d_ws, size_t ws_size,
                              hipStream_t stream) {
  (void)in_sizes; (void)n_in; (void)out_size; (void)ws_size;
  const float* xs      = (const float*)d_in[0];
  const int*   actions = (const int*)d_in[1];
  const float* open_wv = (const float*)d_in[2];
  const float* Wih  = (const float*)d_in[3];
  const float* Whh  = (const float*)d_in[4];
  const float* bih  = (const float*)d_in[5];
  const float* bhh  = (const float*)d_in[6];
  const float* Wfih = (const float*)d_in[7];
  const float* Wfhh = (const float*)d_in[8];
  const float* bfih = (const float*)d_in[9];
  const float* bfhh = (const float*)d_in[10];
  const float* Wbih = (const float*)d_in[11];
  const float* Wbhh = (const float*)d_in[12];
  const float* bbih = (const float*)d_in[13];
  const float* bbhh = (const float*)d_in[14];
  const float* Wc   = (const float*)d_in[15];
  const float* bc   = (const float*)d_in[16];
  char*  ws  = (char*)d_ws;
  float* out = (float*)d_out;

  // 6*262144 + 131072 packed halves, 256 threads each
  pack_weights<<<6656, 256, 0, stream>>>(Wih, Whh, Wfih, Wfhh, Wbih, Wbhh, Wc,
                                         (half_t*)ws);
  prep_bias<<<13, 256, 0, stream>>>(bih, bhh, bfih, bfhh, bbih, bbhh, bc,
                                    (float*)(ws + OFF_BIAS_B));
  stack_lstm_kernel<<<NBLK, BT, 0, stream>>>(xs, actions, open_wv, ws, out);
}
